// MDTA_11879879543502
// MI455X (gfx1250) — matrix operand order repaired and checked by reading, not compiled, not run on hardware
//
#include <hip/hip_runtime.h>

// ---------------------------------------------------------------------------
// MDTA (Restormer transposed channel attention) for MI455X / gfx1250.
// B=2, H=W=256, C=192, heads=4, c=48, N=HW=65536.
//
// Bandwidth-bound (~0.9GB moved vs ~50 GFLOP): intermediates kept in bf16,
// all matrix math on v_wmma_f32_16x16x32_bf16 (f32 accumulate).
//
// Key structure:
//  - stationary GEMM operands (w_qkv, Mcomb) pre-packed into WMMA
//    fragment-native layout [tile][kstep][lane][16]: B fragment = one 32B
//    contiguous per-lane load (2x global_load_b128)
//  - register tiling sized to avoid spills: 12 v8f accumulators (96 VGPRs)
//    per wave + A frag + pipelined B frags < 256 VGPR budget
//  - attention applied as Mcomb = W_out^T * blockdiag(softmax(attn)) folded
//    into a single output GEMM over v
//
// Workspace (assumes ws_size >= ~304 MB).
// ---------------------------------------------------------------------------

typedef __attribute__((ext_vector_type(16))) __bf16 v16bf;
typedef __attribute__((ext_vector_type(2)))  __bf16 v2bf;
typedef __attribute__((ext_vector_type(8)))  float  v8f;

#define NPIX   65536      // H*W
#define CQKV   576
#define CDIM   192
#define HEADS  4
#define CH     48         // per-head channels

// A-fragment (16x32 bf16, MxK) per-lane K index for element e:
// half=lane>>4; K in {0..15} (half 0) / {16..31} (half 1), matching B packing
__device__ __forceinline__ int aK(int e, int half) {
  return (half << 4) + e;
}

__device__ __forceinline__ v8f wmma_bf16(v16bf a, v16bf b, v8f c) {
  return __builtin_amdgcn_wmma_f32_16x16x32_bf16(false, a, false, b,
                                                 (short)0, c, false, false);
}

// ---------------------------------------------------------------------------
// K0: pack w_qkv f32 [192][576] into B-fragment layout
//     wpack[nt(36)][ks(6)][lane(32)][e(16)] bf16,
//     element = w[(ks*32 + 16*half + e)*576 + nt*16 + l15]
// ---------------------------------------------------------------------------
__global__ void pack_wqkv(const float* __restrict__ w, __bf16* __restrict__ wpack) {
  const int t = blockIdx.x * blockDim.x + threadIdx.x;   // [0, 36*6*32)
  const int ks = (t / 32) % 6;
  const int nt = t / (6 * 32);
  const int L = t & 31;
  const int half = L >> 4, l15 = L & 15;
  __bf16* dst = wpack + (size_t)t * 16;
#pragma unroll
  for (int e = 0; e < 16; ++e)
    dst[e] = (__bf16)w[(size_t)(ks * 32 + (half << 4) + e) * CQKV + nt * 16 + l15];
}

// ---------------------------------------------------------------------------
// K1: qkv = x @ W_qkv + b_qkv   (M=131072, K=192, N=576), bf16 out.
// grid (2048, 3), block 128: wave w -> rows (bx*4+w)*16, cols ny*192..+191.
// 12 accumulators/wave (96 VGPRs, no spill); one A fragment -> 12 WMMAs.
// ---------------------------------------------------------------------------
__global__ void __launch_bounds__(128, 1)
qkv_gemm(const float* __restrict__ x, const __bf16* __restrict__ wpack,
         const float* __restrict__ bias, __bf16* __restrict__ out) {
  const int lane = threadIdx.x & 31, wave = threadIdx.x >> 5;
  const int half = lane >> 4, l15 = lane & 15;
  const int m0 = (blockIdx.x * 4 + wave) * 16;
  const int ntb = blockIdx.y * 12;          // N-tile base (of 36)
  const int row = m0 + l15;
  __builtin_prefetch(&x[(size_t)row * CDIM], 0, 0);
  v8f acc[12] = {};
#pragma unroll
  for (int ks = 0; ks < 6; ++ks) {
    v16bf a;
#pragma unroll
    for (int e = 0; e < 16; ++e)
      a[e] = (__bf16)x[(size_t)row * CDIM + ks * 32 + aK(e, half)];
#pragma unroll
    for (int nt = 0; nt < 12; ++nt) {
      const v16bf b = *(const v16bf*)(wpack +
          (((size_t)(ntb + nt) * 6 + ks) * 32 + lane) * 16);
      acc[nt] = wmma_bf16(a, b, acc[nt]);
    }
  }
#pragma unroll
  for (int nt = 0; nt < 12; ++nt) {
    const int col = (ntb + nt) * 16 + l15;
    const float bv = bias[col];
#pragma unroll
    for (int r = 0; r < 8; ++r) {
      const int m = m0 + r + (half << 3);
      out[(size_t)m * CQKV + col] = (__bf16)(acc[nt][r] + bv);
    }
  }
}

// ---------------------------------------------------------------------------
// K2: 3x3 depthwise conv ('SAME', zero pad), bf16 -> bf16, f32 accumulate.
// Two channels per thread (b32 pair loads/stores).
// ---------------------------------------------------------------------------
__global__ void dwconv(const __bf16* __restrict__ in, const float* __restrict__ w,
                       const float* __restrict__ bias, __bf16* __restrict__ out) {
  size_t idx = (size_t)blockIdx.x * blockDim.x + threadIdx.x;
  const size_t totalPairs = (size_t)2 * 256 * 256 * (CQKV / 2);
  if (idx >= totalPairs) return;
  const int ch = (int)(idx % (CQKV / 2)) * 2;
  size_t p = idx / (CQKV / 2);
  const int xx = (int)(p % 256); p /= 256;
  const int yy = (int)(p % 256);
  const int bb = (int)(p / 256);
  float a0 = bias[ch], a1 = bias[ch + 1];
#pragma unroll
  for (int ky = 0; ky < 3; ++ky) {
    const int sy = yy + ky - 1;
    if (sy < 0 || sy > 255) continue;
#pragma unroll
    for (int kx = 0; kx < 3; ++kx) {
      const int sx = xx + kx - 1;
      if (sx < 0 || sx > 255) continue;
      const v2bf v = *(const v2bf*)&in[(((size_t)bb * 256 + sy) * 256 + sx) * CQKV + ch];
      a0 += (float)v[0] * w[(ky * 3 + kx) * CQKV + ch];
      a1 += (float)v[1] * w[(ky * 3 + kx) * CQKV + ch + 1];
    }
  }
  v2bf o; o[0] = (__bf16)a0; o[1] = (__bf16)a1;
  *(v2bf*)&out[(((size_t)bb * 256 + yy) * 256 + xx) * CQKV + ch] = o;
}

// ---------------------------------------------------------------------------
// K3: zero the f32 accumulators (sq: 2*384, Sglob: 2*4*48*48).
// ---------------------------------------------------------------------------
__global__ void zero_acc(float* __restrict__ sq, float* __restrict__ Sglob) {
  const int t = blockIdx.x * blockDim.x + threadIdx.x;
  if (t < 2 * 384) sq[t] = 0.f;
  for (int i = t; i < 2 * HEADS * CH * CH; i += gridDim.x * blockDim.x) Sglob[i] = 0.f;
}

// ---------------------------------------------------------------------------
// K4: per-(b, channel) sum of squares over HW for q (ch 0..191), k (192..383).
// ---------------------------------------------------------------------------
__global__ void sumsq(const __bf16* __restrict__ qkv, float* __restrict__ sq) {
  const int bb = blockIdx.y;
  const int n0 = blockIdx.x * 64;
  for (int ch = threadIdx.x; ch < 2 * CDIM; ch += 256) {
    float local = 0.f;
    for (int n = n0; n < n0 + 64; ++n) {
      const float v = (float)qkv[(((size_t)bb * NPIX) + n) * CQKV + ch];
      local += v * v;
    }
    atomicAdd(&sq[bb * 384 + ch], local);
  }
}

// ---------------------------------------------------------------------------
// K5: split-K raw Gram: Sglob[b,h,cq,ck] += sum_n q[cq,n]*k[ck,n].
// grid (heads, B, 32 K-slices), block 256 (8 waves); 9 WMMA per 6 fragments,
// ds_add_f32 block reduction, global atomic combine.
// ---------------------------------------------------------------------------
__global__ void __launch_bounds__(256, 1)
gram_partial(const __bf16* __restrict__ qkv, float* __restrict__ Sglob) {
  __shared__ float S[CH * CH];
  const int h = blockIdx.x, bb = blockIdx.y, slice = blockIdx.z;
  const int lane = threadIdx.x & 31, wave = threadIdx.x >> 5;
  const int half = lane >> 4, l15 = lane & 15;
  for (int i = threadIdx.x; i < CH * CH; i += 256) S[i] = 0.f;
  __syncthreads();

  const int qb = h * CH;           // q channel base
  const int kb = CDIM + h * CH;    // k channel base
  const size_t base = (size_t)bb * NPIX * CQKV;
  const int nstart = (slice * 8 + wave) * 256;

  v8f acc[3][3] = {};
  for (int n0 = nstart; n0 < nstart + 256; n0 += 32) {
    v16bf aF[3], bF[3];
#pragma unroll
    for (int t = 0; t < 3; ++t) {
#pragma unroll
      for (int e = 0; e < 16; ++e) {
        const int na = n0 + aK(e, half);      // A: M=q-channel, K=n
        aF[t][e] = qkv[base + (size_t)na * CQKV + qb + t * 16 + l15];
        const int nb = n0 + (half << 4) + e;  // B: K=n, N=k-channel
        bF[t][e] = qkv[base + (size_t)nb * CQKV + kb + t * 16 + l15];
      }
    }
#pragma unroll
    for (int mt = 0; mt < 3; ++mt)
#pragma unroll
      for (int nt = 0; nt < 3; ++nt)
        acc[mt][nt] = wmma_bf16(aF[mt], bF[nt], acc[mt][nt]);
  }
#pragma unroll
  for (int mt = 0; mt < 3; ++mt)
#pragma unroll
    for (int nt = 0; nt < 3; ++nt)
#pragma unroll
      for (int r = 0; r < 8; ++r) {
        const int m = mt * 16 + r + (half << 3);
        const int n = nt * 16 + l15;
        atomicAdd(&S[m * CH + n], acc[mt][nt][r]);   // ds_add_f32
      }
  __syncthreads();
  float* dst = Sglob + ((size_t)bb * HEADS + h) * CH * CH;
  for (int i = threadIdx.x; i < CH * CH; i += 256) atomicAdd(&dst[i], S[i]);
}

// ---------------------------------------------------------------------------
// K6: fold L2 norms + temperature, row softmax -> attn.
// ---------------------------------------------------------------------------
__global__ void softmax_k(const float* __restrict__ Sglob, const float* __restrict__ sq,
                          const float* __restrict__ temp, float* __restrict__ attn) {
  const int h = blockIdx.x, bb = blockIdx.y;
  const int r = threadIdx.x;
  if (r >= CH) return;
  const float* S = Sglob + ((size_t)bb * HEADS + h) * CH * CH;
  const float qs = 1.f / fmaxf(sqrtf(sq[bb * 384 + h * CH + r]), 1e-12f);
  const float T = temp[h];
  float row[CH];
  float mx = -3.4e38f;
#pragma unroll
  for (int d = 0; d < CH; ++d) {
    const float ks = 1.f / fmaxf(sqrtf(sq[bb * 384 + CDIM + h * CH + d]), 1e-12f);
    const float s = S[r * CH + d] * qs * ks * T;
    row[d] = s;
    mx = fmaxf(mx, s);
  }
  float sum = 0.f;
#pragma unroll
  for (int d = 0; d < CH; ++d) { row[d] = __expf(row[d] - mx); sum += row[d]; }
  const float inv = 1.f / sum;
  float* dst = attn + (((size_t)bb * HEADS + h) * CH + r) * CH;
#pragma unroll
  for (int d = 0; d < CH; ++d) dst[d] = row[d] * inv;
}

// ---------------------------------------------------------------------------
// K7: Mcf[b][dg][co] = sum_cl w_out[h*48+cl, co] * attn[b,h,cl,d]
// (W_out^T * blockdiag(attn)); f32, tiny.
// ---------------------------------------------------------------------------
__global__ void mcomb(const float* __restrict__ attn, const float* __restrict__ wout,
                      float* __restrict__ Mcf) {
  const int bb = blockIdx.x;
  const int co = threadIdx.x;
  for (int dg = 0; dg < CDIM; ++dg) {
    const int h = dg / CH, d = dg % CH;
    float s = 0.f;
    for (int cl = 0; cl < CH; ++cl)
      s += wout[(h * CH + cl) * CDIM + co] *
           attn[(((size_t)bb * HEADS + h) * CH + cl) * CH + d];
    Mcf[((size_t)bb * CDIM + dg) * CDIM + co] = s;
  }
}

// ---------------------------------------------------------------------------
// K8: pack Mcf into B-fragment layout Mcpack[b][nt(12)][ks(6)][lane][e] bf16.
// ---------------------------------------------------------------------------
__global__ void pack_mc(const float* __restrict__ Mcf, __bf16* __restrict__ mpack) {
  const int t = blockIdx.x * blockDim.x + threadIdx.x;   // [0, 2*12*6*32)
  const int bb = t / (12 * 6 * 32);
  const int rem = t % (12 * 6 * 32);
  const int nt = rem / (6 * 32);
  const int ks = (rem / 32) % 6;
  const int L = t & 31;
  const int half = L >> 4, l15 = L & 15;
  __bf16* dst = mpack + (size_t)t * 16;
#pragma unroll
  for (int e = 0; e < 16; ++e)
    dst[e] = (__bf16)Mcf[((size_t)bb * CDIM + ks * 32 + (half << 4) + e) * CDIM +
                         nt * 16 + l15];
}

// ---------------------------------------------------------------------------
// K9: out[b,n,co] = sum_dg v[b,n,dg] * Mcomb[b][dg][co] + b_out[co]
// grid (1024, 2), block 128: wave owns 16 rows x all 192 cols (12 accums).
// v = qkv_dw channels 384..575 (bf16). One A fragment feeds 12 WMMAs.
// ---------------------------------------------------------------------------
__global__ void __launch_bounds__(128, 1)
out_gemm(const __bf16* __restrict__ qkv, const __bf16* __restrict__ mpack,
         const float* __restrict__ bias, float* __restrict__ out) {
  const int lane = threadIdx.x & 31, wave = threadIdx.x >> 5;
  const int half = lane >> 4, l15 = lane & 15;
  const int bb = blockIdx.y;
  const int m0 = (blockIdx.x * 4 + wave) * 16;
  const size_t base = (size_t)bb * NPIX * CQKV;
  const __bf16* mp = mpack + (size_t)bb * 12 * 6 * 32 * 16;
  const int row = m0 + l15;
  __builtin_prefetch(&qkv[base + (size_t)row * CQKV + 2 * CDIM], 0, 0);
  v8f acc[12] = {};
#pragma unroll
  for (int ks = 0; ks < 6; ++ks) {
    v16bf a;
#pragma unroll
    for (int e = 0; e < 16; ++e)
      a[e] = qkv[base + (size_t)row * CQKV + 2 * CDIM + ks * 32 + aK(e, half)];
#pragma unroll
    for (int nt = 0; nt < 12; ++nt) {
      const v16bf b = *(const v16bf*)(mp + (((size_t)nt * 6 + ks) * 32 + lane) * 16);
      acc[nt] = wmma_bf16(a, b, acc[nt]);
    }
  }
#pragma unroll
  for (int nt = 0; nt < 12; ++nt) {
    const int col = nt * 16 + l15;
    const float bv = bias[col];
#pragma unroll
    for (int r = 0; r < 8; ++r)
      out[((size_t)bb * NPIX + m0 + r + (half << 3)) * CDIM + col] = acc[nt][r] + bv;
  }
}

// ---------------------------------------------------------------------------
extern "C" void kernel_launch(void* const* d_in, const int* in_sizes, int n_in,
                              void* d_out, int out_size, void* d_ws, size_t ws_size,
                              hipStream_t stream) {
  const float* x     = (const float*)d_in[0];
  const float* w_qkv = (const float*)d_in[1];
  const float* b_qkv = (const float*)d_in[2];
  const float* w_dw  = (const float*)d_in[3];
  const float* b_dw  = (const float*)d_in[4];
  const float* temp  = (const float*)d_in[5];
  const float* w_out = (const float*)d_in[6];
  const float* b_out = (const float*)d_in[7];
  float* out = (float*)d_out;

  const size_t qkv_elems = (size_t)2 * NPIX * CQKV;
  char* ws = (char*)d_ws;
  size_t off = 0;
  auto take = [&](size_t bytes) {
    char* p = ws + off;
    off = (off + bytes + 255) & ~(size_t)255;
    return p;
  };
  __bf16* qkv_raw = (__bf16*)take(qkv_elems * sizeof(__bf16));
  __bf16* qkv_dw  = (__bf16*)take(qkv_elems * sizeof(__bf16));
  __bf16* wpack   = (__bf16*)take((size_t)36 * 6 * 32 * 16 * sizeof(__bf16));
  float*  sq      = (float*) take((size_t)2 * 384 * sizeof(float));
  float*  Sglob   = (float*) take((size_t)2 * HEADS * CH * CH * sizeof(float));
  float*  attn    = (float*) take((size_t)2 * HEADS * CH * CH * sizeof(float));
  float*  Mcf     = (float*) take((size_t)2 * CDIM * CDIM * sizeof(float));
  __bf16* Mcpack  = (__bf16*)take((size_t)2 * 12 * 6 * 32 * 16 * sizeof(__bf16));
  (void)ws_size; (void)in_sizes; (void)n_in; (void)out_size;

  // 0) pack W_qkv into fragment layout
  pack_wqkv<<<27, 256, 0, stream>>>(w_qkv, wpack);               // 36*6*32 = 6912
  // 1) qkv projection GEMM (WMMA bf16, 12 acc tiles/wave, spill-free)
  qkv_gemm<<<dim3(2048, 3), 128, 0, stream>>>(x, wpack, b_qkv, qkv_raw);
  // 2) 3x3 depthwise conv (pairwise channels)
  dwconv<<<(unsigned)((qkv_elems / 2 + 255) / 256), 256, 0, stream>>>(
      qkv_raw, w_dw, b_dw, qkv_dw);
  // 3) zero reduction accumulators
  zero_acc<<<80, 256, 0, stream>>>(sq, Sglob);
  // 4) per-channel sum of squares (q & k)
  sumsq<<<dim3(NPIX / 64, 2), 256, 0, stream>>>(qkv_dw, sq);
  // 5) split-K Gram matrices via WMMA + LDS/global atomic reduction
  gram_partial<<<dim3(HEADS, 2, 32), 256, 0, stream>>>(qkv_dw, Sglob);
  // 6) normalize + temperature + softmax
  softmax_k<<<dim3(HEADS, 2), 64, 0, stream>>>(Sglob, sq, temp, attn);
  // 7) fold attention into output projection: Mcf = W_out^T * blockdiag(attn)
  mcomb<<<2, CDIM, 0, stream>>>(attn, w_out, Mcf);
  // 8) pack Mcf into fragment layout
  pack_mc<<<18, 256, 0, stream>>>(Mcf, Mcpack);                  // 2*12*6*32 = 4608
  // 9) fused attention-apply + output projection GEMM (WMMA bf16, 12 acc/wave)
  out_gemm<<<dim3(1024, 2), 128, 0, stream>>>(qkv_dw, Mcpack, b_out, out);
}
